// GCSA_5884105195632
// MI455X (gfx1250) — matrix operand order repaired and checked by reading, not compiled, not run on hardware
//
#include <hip/hip_runtime.h>
#include <hip/hip_bf16.h>

// ---------------------------------------------------------------------------
// CDNA5 (gfx1250) implementation of the dual-key gated channel-attention block.
// All GEMM stages run on v_wmma_f32_16x16x32_bf16 (fp32 accumulate); norms,
// dilated depthwise conv, softmax and the gated fusion stay in fp32 VALU.
// GEMM B-operands are staged through double-buffered LDS (transposed so that
// fragment reads are contiguous b128 ds loads); global loads for slab i+1 are
// issued before the WMMAs of slab i so VMEM latency hides under matrix math.
// ---------------------------------------------------------------------------

typedef __attribute__((ext_vector_type(16))) __bf16 v16bf;
typedef __attribute__((ext_vector_type(8)))  __bf16 v8bf;
typedef __attribute__((ext_vector_type(8)))  float  v8f;

union FragBF { v16bf v; v8bf h[2]; };

__device__ __forceinline__ __bf16 f2bf(float f) {
  union { float f; unsigned u; } c; c.f = f;
  unsigned u = c.u;
  unsigned r = (u + 0x7FFFu + ((u >> 16) & 1u)) >> 16;   // round-nearest-even
  union { unsigned short s; __bf16 b; } o; o.s = (unsigned short)r;
  return o.b;
}
__device__ __forceinline__ float bf2f(__bf16 b) {
  union { unsigned short s; __bf16 b; } i; i.b = b;
  union { unsigned u; float f; } o; o.u = ((unsigned)i.s) << 16;
  return o.f;
}

__device__ __forceinline__ v8f wmma_bf16(const FragBF& a, const FragBF& b, v8f c) {
  return __builtin_amdgcn_wmma_f32_16x16x32_bf16(
      false, a.v, false, b.v, (short)0, c, false, false);
}

// ---------------------------------------------------------------------------
// fp32 -> bf16 conversion (grid-stride)
// ---------------------------------------------------------------------------
__global__ void cvt_f32_bf16(const float* __restrict__ src,
                             __bf16* __restrict__ dst, long n) {
  long i = (long)blockIdx.x * blockDim.x + threadIdx.x;
  long stride = (long)gridDim.x * blockDim.x;
  for (; i < n; i += stride) dst[i] = f2bf(src[i]);
}

// ---------------------------------------------------------------------------
// Generic bf16 WMMA GEMM:  D[z] (MxN, row-major) = A[z] (MxK) * B[z] (KxN)
// A is K-contiguous (direct global fragments); B staged through double-
// buffered transposed LDS tiles. Block: 256 threads = 8 waves; block tile
// M=128, N=64; K step 32. Assumes M%128==0, N%64==0, K%32==0 (all call sites).
// ---------------------------------------------------------------------------
__global__ void __launch_bounds__(256) wmma_gemm_bf16(
    const __bf16* __restrict__ A, const __bf16* __restrict__ B,
    void* __restrict__ Dp, int M, int N, int K,
    long aBatch, long bBatch, long dBatch, int storeBf16) {
  __shared__ __bf16 Bt[2][64 * 48];             // 64 n-rows, K-pitch 48 (padded)
  const int tid  = threadIdx.x;
  const int wave = tid >> 5, lane = tid & 31;
  const int z = blockIdx.z;
  const int mBase = blockIdx.y * 128 + wave * 16;
  const int nBase = blockIdx.x * 64;
  A += (long)z * aBatch;
  B += (long)z * bBatch;

  v8f acc[4] = {{}, {}, {}, {}};
  const int kk    = tid >> 3;                   // 0..31 (k within slab)
  const int n8    = (tid & 7) * 8;              // 0..56 (n chunk)
  const int lcol  = lane & 15;
  const int khalf = lane >> 4;
  const int kb    = khalf * 8;                  // B-fragment k base per ISA layout
  const long aRow = (long)(mBase + lcol) * K;
  const __bf16* bSlab = B + (long)kk * N + nBase + n8;

  // prologue: stage slab 0 into buffer 0
  {
    v8bf bv = *(const v8bf*)(bSlab);
#pragma unroll
    for (int j = 0; j < 8; ++j) Bt[0][(n8 + j) * 48 + kk] = bv[j];
  }
  __syncthreads();

  int buf = 0;
  for (int k0 = 0; k0 < K; k0 += 32, buf ^= 1) {
    const bool more = (k0 + 32 < K);
    v8bf bnext = {};
    if (more) {                                 // global load for slab i+1 first
      bnext = *(const v8bf*)(bSlab + (long)(k0 + 32) * N);
      if (k0 + 64 < K)
        __builtin_prefetch(bSlab + (long)(k0 + 64) * N, 0, 1);
    }
    // A fragment: ISA 16-bit A layout (K = khalf*8..+7 and 16+khalf*8..+7)
    FragBF a;
    a.h[0] = *(const v8bf*)(A + aRow + k0 + khalf * 8);
    a.h[1] = *(const v8bf*)(A + aRow + k0 + 16 + khalf * 8);
    // preload all four B fragments, then issue the WMMAs back-to-back
    const __bf16* bt = &Bt[buf][lcol * 48 + kb];
    FragBF b0, b1, b2, b3;
    b0.h[0] = *(const v8bf*)(bt + 0 * 768); b0.h[1] = *(const v8bf*)(bt + 0 * 768 + 16);
    b1.h[0] = *(const v8bf*)(bt + 1 * 768); b1.h[1] = *(const v8bf*)(bt + 1 * 768 + 16);
    b2.h[0] = *(const v8bf*)(bt + 2 * 768); b2.h[1] = *(const v8bf*)(bt + 2 * 768 + 16);
    b3.h[0] = *(const v8bf*)(bt + 3 * 768); b3.h[1] = *(const v8bf*)(bt + 3 * 768 + 16);
    acc[0] = wmma_bf16(a, b0, acc[0]);
    acc[1] = wmma_bf16(a, b1, acc[1]);
    acc[2] = wmma_bf16(a, b2, acc[2]);
    acc[3] = wmma_bf16(a, b3, acc[3]);
    if (more) {                                 // stage slab i+1 into other buffer
#pragma unroll
      for (int j = 0; j < 8; ++j) Bt[buf ^ 1][(n8 + j) * 48 + kk] = bnext[j];
    }
    __syncthreads();                            // single barrier per K-step
  }

  // epilogue: C/D layout VGPR r -> M = r + 8*(lane>=16), N = lane&15
  if (storeBf16) {
    __bf16* D = (__bf16*)Dp + (long)z * dBatch;
#pragma unroll
    for (int r = 0; r < 8; ++r) {
      long base = (long)(mBase + r + khalf * 8) * N + nBase + lcol;
#pragma unroll
      for (int t = 0; t < 4; ++t) D[base + t * 16] = f2bf(acc[t][r]);
    }
  } else {
    float* D = (float*)Dp + (long)z * dBatch;
#pragma unroll
    for (int r = 0; r < 8; ++r) {
      long base = (long)(mBase + r + khalf * 8) * N + nBase + lcol;
#pragma unroll
      for (int t = 0; t < 4; ++t) D[base + t * 16] = acc[t][r];
    }
  }
}

// ---------------------------------------------------------------------------
// Depthwise 3x3, dilation 2, padding 2 over 64x64 images (bf16 in/out, fp32
// math) + reciprocal L2 norms for the q (ch<256) and k (256<=ch<512) planes.
// One block per (batch, channel) row.
// ---------------------------------------------------------------------------
__global__ void __launch_bounds__(256) dwconv_rnorm(
    const __bf16* __restrict__ in, const float* __restrict__ wdw,
    __bf16* __restrict__ outp, float* __restrict__ rnq, float* __restrict__ rnk) {
  __shared__ float red[256];
  const int tid = threadIdx.x;
  const int row = blockIdx.x;                   // b*768 + ch
  const int b  = row / 768;
  const int ch = row - b * 768;
  const __bf16* src = in + (long)row * 4096;
  __bf16* dst = outp + (long)row * 4096;
  float w[9];
#pragma unroll
  for (int i = 0; i < 9; ++i) w[i] = wdw[ch * 9 + i];
  float ss = 0.f;
  for (int p = tid; p < 4096; p += 256) {
    int y = p >> 6, x = p & 63;
    float s = 0.f;
#pragma unroll
    for (int ky = 0; ky < 3; ++ky) {
      int yy = y + (ky - 1) * 2;
      if (yy < 0 || yy > 63) continue;
#pragma unroll
      for (int kx = 0; kx < 3; ++kx) {
        int xx = x + (kx - 1) * 2;
        if (xx < 0 || xx > 63) continue;
        s += w[ky * 3 + kx] * bf2f(src[yy * 64 + xx]);
      }
    }
    dst[p] = f2bf(s);
    ss += s * s;
  }
  if (ch < 512) {                               // q / k planes need norms
    red[tid] = ss; __syncthreads();
    for (int off = 128; off > 0; off >>= 1) {
      if (tid < off) red[tid] += red[tid + off];
      __syncthreads();
    }
    if (tid == 0) {
      float rn = 1.f / fmaxf(sqrtf(red[0]), 1e-12f);
      if (ch < 256) rnq[b * 256 + ch] = rn;
      else          rnk[b * 256 + (ch - 256)] = rn;
    }
  }
}

// ---------------------------------------------------------------------------
// Reciprocal row L2 norm of a bf16 matrix (one block per row).
// ---------------------------------------------------------------------------
__global__ void __launch_bounds__(256) row_rnorm_bf16(
    const __bf16* __restrict__ m, float* __restrict__ rn, int ncol) {
  __shared__ float red[256];
  const int row = blockIdx.x, tid = threadIdx.x;
  const __bf16* src = m + (long)row * ncol;
  float ss = 0.f;
  for (int p = tid; p < ncol; p += 256) { float v = bf2f(src[p]); ss += v * v; }
  red[tid] = ss; __syncthreads();
  for (int off = 128; off > 0; off >>= 1) {
    if (tid < off) red[tid] += red[tid + off];
    __syncthreads();
  }
  if (tid == 0) rn[row] = 1.f / fmaxf(sqrtf(red[0]), 1e-12f);
}

// ---------------------------------------------------------------------------
// Attention logits: attn[b,c,d] = rnq_c*( <q_c,k_d>*rnk_d + <q_c,ck_d>*rnc_d )
// Both operands are K(spatial=4096)-contiguous rows -> pure global-fragment
// WMMA, no LDS needed. One wave per 16x16 output tile; K unrolled x2 so four
// WMMAs issue per iteration over batched b128 loads.
// ---------------------------------------------------------------------------
__global__ void __launch_bounds__(32) attn_wmma(
    const __bf16* __restrict__ qkv, const __bf16* __restrict__ cnnk,
    const float* __restrict__ rnq, const float* __restrict__ rnk,
    const float* __restrict__ rnc, float* __restrict__ attn) {
  const int lane = threadIdx.x & 31;
  const int b  = blockIdx.z;
  const int c0 = blockIdx.y * 16, d0 = blockIdx.x * 16;
  const __bf16* q  = qkv + (long)b * 768 * 4096;        // channels [0,256)
  const __bf16* k  = q + 256 * 4096;                    // channels [256,512)
  const __bf16* ck = cnnk + (long)b * 256 * 4096;
  const int lcol  = lane & 15;
  const int khalf = lane >> 4;
  const long aRow = (long)(c0 + lcol) * 4096;
  const long bRow = (long)(d0 + lcol) * 4096;
  const int kb = khalf * 8;
  v8f acc1 = {}, acc2 = {};
  for (int k0 = 0; k0 < 4096; k0 += 64) {
    __builtin_prefetch(q  + aRow + k0 + 128, 0, 1);
    __builtin_prefetch(k  + bRow + k0 + 128, 0, 1);
    __builtin_prefetch(ck + bRow + k0 + 128, 0, 1);
    FragBF a0, a1, bk0, bk1, bc0, bc1;
    a0.h[0]  = *(const v8bf*)(q  + aRow + k0 + khalf * 8);
    a0.h[1]  = *(const v8bf*)(q  + aRow + k0 + 16 + khalf * 8);
    a1.h[0]  = *(const v8bf*)(q  + aRow + k0 + 32 + khalf * 8);
    a1.h[1]  = *(const v8bf*)(q  + aRow + k0 + 48 + khalf * 8);
    bk0.h[0] = *(const v8bf*)(k  + bRow + k0 + kb);
    bk0.h[1] = *(const v8bf*)(k  + bRow + k0 + kb + 16);
    bk1.h[0] = *(const v8bf*)(k  + bRow + k0 + 32 + kb);
    bk1.h[1] = *(const v8bf*)(k  + bRow + k0 + 32 + kb + 16);
    bc0.h[0] = *(const v8bf*)(ck + bRow + k0 + kb);
    bc0.h[1] = *(const v8bf*)(ck + bRow + k0 + kb + 16);
    bc1.h[0] = *(const v8bf*)(ck + bRow + k0 + 32 + kb);
    bc1.h[1] = *(const v8bf*)(ck + bRow + k0 + 32 + kb + 16);
    acc1 = wmma_bf16(a0, bk0, acc1);
    acc2 = wmma_bf16(a0, bc0, acc2);
    acc1 = wmma_bf16(a1, bk1, acc1);
    acc2 = wmma_bf16(a1, bc1, acc2);
  }
  const float* rq = rnq + b * 256;
  const float* rk = rnk + b * 256;
  const float* rc = rnc + b * 256;
#pragma unroll
  for (int r = 0; r < 8; ++r) {
    int cc = c0 + r + khalf * 8;
    int dd = d0 + lcol;
    attn[((long)b * 256 + cc) * 256 + dd] =
        rq[cc] * (acc1[r] * rk[dd] + acc2[r] * rc[dd]);
  }
}

// ---------------------------------------------------------------------------
// Row softmax over 256 logits (fp32 in, bf16 out). One block per (b,c) row.
// ---------------------------------------------------------------------------
__global__ void __launch_bounds__(256) softmax_row256(
    const float* __restrict__ a, __bf16* __restrict__ o) {
  __shared__ float red[256];
  const int row = blockIdx.x, tid = threadIdx.x;
  float v = a[(long)row * 256 + tid];
  red[tid] = v; __syncthreads();
  for (int off = 128; off > 0; off >>= 1) {
    if (tid < off) red[tid] = fmaxf(red[tid], red[tid + off]);
    __syncthreads();
  }
  float mx = red[0]; __syncthreads();
  float e = __expf(v - mx);
  red[tid] = e; __syncthreads();
  for (int off = 128; off > 0; off >>= 1) {
    if (tid < off) red[tid] += red[tid + off];
    __syncthreads();
  }
  o[(long)row * 256 + tid] = f2bf(e / red[0]);
}

// ---------------------------------------------------------------------------
// Final stage: o1 = Wp1 @ out, o2 = Wp2 @ out (shared double-buffered LDS B
// tile, dual WMMA accumulators), then branch-softmax gating with x/x2.
// ---------------------------------------------------------------------------
__global__ void __launch_bounds__(256) final_gemm_gate(
    const __bf16* __restrict__ W1, const __bf16* __restrict__ W2,
    const __bf16* __restrict__ Bm, const float* __restrict__ x,
    const float* __restrict__ x2, float* __restrict__ out) {
  __shared__ __bf16 Bt[2][64 * 48];
  const int tid  = threadIdx.x;
  const int wave = tid >> 5, lane = tid & 31;
  const int z = blockIdx.z;
  const int mBase = blockIdx.y * 128 + wave * 16;
  const int nBase = blockIdx.x * 64;
  const __bf16* B = Bm + (long)z * 256 * 4096;

  v8f a1[4] = {{}, {}, {}, {}};
  v8f a2[4] = {{}, {}, {}, {}};
  const int kk    = tid >> 3;
  const int n8    = (tid & 7) * 8;
  const int lcol  = lane & 15;
  const int khalf = lane >> 4;
  const int kb    = khalf * 8;
  const long aRow = (long)(mBase + lcol) * 256;
  const __bf16* bSlab = B + (long)kk * 4096 + nBase + n8;

  {
    v8bf bv = *(const v8bf*)(bSlab);
#pragma unroll
    for (int j = 0; j < 8; ++j) Bt[0][(n8 + j) * 48 + kk] = bv[j];
  }
  __syncthreads();

  int buf = 0;
  for (int k0 = 0; k0 < 256; k0 += 32, buf ^= 1) {
    const bool more = (k0 + 32 < 256);
    v8bf bnext = {};
    if (more) {
      bnext = *(const v8bf*)(bSlab + (long)(k0 + 32) * 4096);
      if (k0 + 64 < 256)
        __builtin_prefetch(bSlab + (long)(k0 + 64) * 4096, 0, 1);
    }
    FragBF f1, f2;
    f1.h[0] = *(const v8bf*)(W1 + aRow + k0 + khalf * 8);
    f1.h[1] = *(const v8bf*)(W1 + aRow + k0 + 16 + khalf * 8);
    f2.h[0] = *(const v8bf*)(W2 + aRow + k0 + khalf * 8);
    f2.h[1] = *(const v8bf*)(W2 + aRow + k0 + 16 + khalf * 8);
    const __bf16* bt = &Bt[buf][lcol * 48 + kb];
    FragBF b0, b1, b2, b3;
    b0.h[0] = *(const v8bf*)(bt + 0 * 768); b0.h[1] = *(const v8bf*)(bt + 0 * 768 + 16);
    b1.h[0] = *(const v8bf*)(bt + 1 * 768); b1.h[1] = *(const v8bf*)(bt + 1 * 768 + 16);
    b2.h[0] = *(const v8bf*)(bt + 2 * 768); b2.h[1] = *(const v8bf*)(bt + 2 * 768 + 16);
    b3.h[0] = *(const v8bf*)(bt + 3 * 768); b3.h[1] = *(const v8bf*)(bt + 3 * 768 + 16);
    a1[0] = wmma_bf16(f1, b0, a1[0]);
    a2[0] = wmma_bf16(f2, b0, a2[0]);
    a1[1] = wmma_bf16(f1, b1, a1[1]);
    a2[1] = wmma_bf16(f2, b1, a2[1]);
    a1[2] = wmma_bf16(f1, b2, a1[2]);
    a2[2] = wmma_bf16(f2, b2, a2[2]);
    a1[3] = wmma_bf16(f1, b3, a1[3]);
    a2[3] = wmma_bf16(f2, b3, a2[3]);
    if (more) {
#pragma unroll
      for (int j = 0; j < 8; ++j) Bt[buf ^ 1][(n8 + j) * 48 + kk] = bnext[j];
    }
    __syncthreads();
  }

#pragma unroll
  for (int r = 0; r < 8; ++r) {
    int mm = mBase + r + khalf * 8;
#pragma unroll
    for (int t = 0; t < 4; ++t) {
      int nn = nBase + t * 16 + lcol;
      float o1 = a1[t][r], o2 = a2[t][r];
      float mx = fmaxf(o1, o2);
      float e1 = __expf(o1 - mx), e2 = __expf(o2 - mx);
      float inv = 1.f / (e1 + e2);
      long idx = ((long)z * 256 + mm) * 4096 + nn;
      out[idx] = (x[idx] * e1 + x2[idx] * e2) * inv;
    }
  }
}

// ---------------------------------------------------------------------------
// Host orchestration
// ---------------------------------------------------------------------------
extern "C" void kernel_launch(void* const* d_in, const int* in_sizes, int n_in,
                              void* d_out, int out_size, void* d_ws,
                              size_t ws_size, hipStream_t stream) {
  (void)in_sizes; (void)n_in; (void)out_size;
  const float* x    = (const float*)d_in[0];
  const float* x2   = (const float*)d_in[1];
  const float* wqkv = (const float*)d_in[2];
  const float* wck  = (const float*)d_in[3];
  const float* wdw  = (const float*)d_in[4];
  const float* wp1  = (const float*)d_in[5];
  const float* wp2  = (const float*)d_in[6];
  float* out = (float*)d_out;

  const long SZA = 16L * 256 * 4096 * 2;        // one bf16 activation set: 32 MiB
  const size_t NEED = (size_t)(8 * SZA) + 786432 + 49152;
  if (ws_size < NEED) return;

  char* ws = (char*)d_ws;
  __bf16* xb    = (__bf16*)(ws);                // x bf16; later aliased by cnn_k
  __bf16* x2b   = (__bf16*)(ws + SZA);          // x2 bf16; later aliased by attn
  __bf16* qraw  = (__bf16*)(ws + 2 * SZA);      // (16,768,4096); later attn@v out
  __bf16* qdw   = (__bf16*)(ws + 5 * SZA);      // (16,768,4096) post-dwconv
  char*   wsw   = ws + 8 * SZA;
  __bf16* wqkvb = (__bf16*)(wsw);
  __bf16* wckb  = (__bf16*)(wsw + 393216);
  __bf16* wp1b  = (__bf16*)(wsw + 393216 + 131072);
  __bf16* wp2b  = (__bf16*)(wsw + 393216 + 2 * 131072);
  float*  rnq   = (float*)(wsw + 786432);
  float*  rnk   = rnq + 16 * 256;
  float*  rnc   = rnk + 16 * 256;
  // aliases (lifetimes verified): cnn_k over xb, attn over x2b, out over qraw
  __bf16* cnnk  = xb;
  float*  attnF = (float*)x2b;                  // (16,256,256) f32
  __bf16* attnB = (__bf16*)(ws + SZA + 4194304);
  __bf16* outbf = qraw;                         // (16,256,4096) bf16
  const __bf16* vmat = qdw + 512 * 4096;        // v planes; batch stride 768*4096

  // 1) fp32 -> bf16
  cvt_f32_bf16<<<2048, 256, 0, stream>>>(x,  xb,  16L * 256 * 4096);
  cvt_f32_bf16<<<2048, 256, 0, stream>>>(x2, x2b, 16L * 256 * 4096);
  cvt_f32_bf16<<<64, 256, 0, stream>>>(wqkv, wqkvb, 768L * 256);
  cvt_f32_bf16<<<32, 256, 0, stream>>>(wck,  wckb,  256L * 256);
  cvt_f32_bf16<<<32, 256, 0, stream>>>(wp1,  wp1b,  256L * 256);
  cvt_f32_bf16<<<32, 256, 0, stream>>>(wp2,  wp2b,  256L * 256);
  // 2) qkv = Wqkv @ x   (768x4096 per batch, K=256)
  wmma_gemm_bf16<<<dim3(64, 6, 16), 256, 0, stream>>>(
      wqkvb, xb, qraw, 768, 4096, 256, 0L, 1048576L, 3145728L, 1);
  // 3) dilated depthwise conv + q/k reciprocal norms
  dwconv_rnorm<<<16 * 768, 256, 0, stream>>>(qraw, wdw, qdw, rnq, rnk);
  // 4) cnn_k = Wck @ x2
  wmma_gemm_bf16<<<dim3(64, 2, 16), 256, 0, stream>>>(
      wckb, x2b, cnnk, 256, 4096, 256, 0L, 1048576L, 1048576L, 1);
  // 5) cnn_k reciprocal norms
  row_rnorm_bf16<<<16 * 256, 256, 0, stream>>>(cnnk, rnc, 4096);
  // 6) attention logits (K=4096, dual-key, norm scales folded in)
  attn_wmma<<<dim3(16, 16, 16), 32, 0, stream>>>(qdw, cnnk, rnq, rnk, rnc, attnF);
  // 7) row softmax -> bf16 probabilities
  softmax_row256<<<16 * 256, 256, 0, stream>>>(attnF, attnB);
  // 8) out = attn @ v   (K=256)
  wmma_gemm_bf16<<<dim3(64, 2, 16), 256, 0, stream>>>(
      attnB, vmat, outbf, 256, 4096, 256, 65536L, 3145728L, 1048576L, 1);
  // 9) p1/p2 projections + branch-softmax gated fusion with x/x2
  final_gemm_gate<<<dim3(64, 2, 16), 256, 0, stream>>>(
      wp1b, wp2b, outbf, x, x2, out);
}
